// RGCN_59115929862915
// MI455X (gfx1250) — matrix operand order repaired and checked by reading, not compiled, not run on hardware
//
#include <hip/hip_runtime.h>

#define FEATS   128
#define N_REL   3

// ---- WMMA vector types (wave32, gfx1250) ----
typedef __attribute__((ext_vector_type(16))) __bf16 v16bf;
typedef __attribute__((ext_vector_type(8)))  float  v8f;

// ---------------------------------------------------------------------------
// 0) zero the output (atomics accumulate into it)
// ---------------------------------------------------------------------------
__global__ void rgcn_zero_out(float4* __restrict__ p, int n4) {
    int i = blockIdx.x * blockDim.x + threadIdx.x;
    if (i < n4) p[i] = make_float4(0.f, 0.f, 0.f, 0.f);
}

// ---------------------------------------------------------------------------
// 1) W[3][128][128] f32 -> bf16 hi/lo, pre-swizzled into WMMA B-fragment
//    layout (ISA 7.12.2, 16-bit B 32x16): lane = kgroup*16 + n, K = (h/8)*16 + kgroup*8 + (h%8)
//    idx = (((r*8 + colTile)*4 + kChunk)*32 + lane)*16 + h
// ---------------------------------------------------------------------------
__global__ void rgcn_prep_w(const float* __restrict__ W,
                            unsigned short* __restrict__ Whi,
                            unsigned short* __restrict__ Wlo) {
    int tid = blockIdx.x * blockDim.x + threadIdx.x;     // over 3*128*128
    if (tid >= N_REL * FEATS * FEATS) return;
    int r = tid >> 14;
    int k = (tid >> 7) & 127;
    int o = tid & 127;
    float x = W[tid];
    __bf16 hi = (__bf16)x;
    __bf16 lo = (__bf16)(x - (float)hi);
    int t  = o >> 4, n = o & 15;
    int c  = k >> 5, kk = k & 31;
    int kg = (kk >> 3) & 1, h = (kk >> 4) * 8 + (kk & 7);
    int lane = kg * 16 + n;
    int idx  = (((r * 8 + t) * 4 + c) * 32 + lane) * 16 + h;
    ((__bf16*)Whi)[idx] = hi;
    ((__bf16*)Wlo)[idx] = lo;
}

// ---------------------------------------------------------------------------
// 1b) X[N][128] f32 -> bf16 hi/lo pre-swizzled into WMMA A-fragment layout
//     (ISA 7.12.2, 16-bit A 16x32): lane L: m = L%16, kgroup = L/16,
//     half h -> K = (h/8)*16 + kgroup*8 + (h%8).
//     Inverse for kk = K%32:  kgroup = (kk>>3)&1 ; h = (kk>>4)*8 + (kk&7)
//     idx = ((mtile*4 + kChunk)*32 + lane)*16 + h
// ---------------------------------------------------------------------------
__global__ void rgcn_prep_x(const float* __restrict__ X,
                            unsigned short* __restrict__ Xhi,
                            unsigned short* __restrict__ Xlo, int n_elems) {
    int tid = blockIdx.x * blockDim.x + threadIdx.x;     // over N*128
    if (tid >= n_elems) return;
    int node = tid >> 7;
    int k    = tid & 127;
    float x = X[tid];
    __bf16 hi = (__bf16)x;
    __bf16 lo = (__bf16)(x - (float)hi);
    int mt = node >> 4, mloc = node & 15;
    int c  = k >> 5,    kk   = k & 31;
    int kg = (kk >> 3) & 1;
    int h  = (kk >> 4) * 8 + (kk & 7);
    int lane = kg * 16 + mloc;
    size_t idx = (((size_t)mt * 4 + c) * 32 + lane) * 16 + h;
    ((__bf16*)Xhi)[idx] = hi;
    ((__bf16*)Xlo)[idx] = lo;
}

// ---------------------------------------------------------------------------
// 2a) FAST GEMM: both operands pre-split/pre-swizzled -> inner loop is just
//     coalesced b128 fragment loads + 3 WMMAs per 32-wide k-chunk.
//     block = 16 rows x 128 cols x 1 relation; wave w -> cols [16w, 16w+16)
// ---------------------------------------------------------------------------
__global__ void __launch_bounds__(256)
rgcn_gemm_pre(const unsigned short* __restrict__ Xhi,
              const unsigned short* __restrict__ Xlo,
              const unsigned short* __restrict__ Whi,
              const unsigned short* __restrict__ Wlo,
              float* __restrict__ HR, int n_nodes) {
    const int lane = threadIdx.x & 31;
    const int w    = threadIdx.x >> 5;
    const int r    = blockIdx.y;
    const int mt   = blockIdx.x;

    const __bf16* ah = (const __bf16*)Xhi + ((size_t)mt * 4) * 512 + lane * 16;
    const __bf16* al = (const __bf16*)Xlo + ((size_t)mt * 4) * 512 + lane * 16;
    const __bf16* bh = (const __bf16*)Whi + ((size_t)(r * 8 + w) * 4) * 512 + lane * 16;
    const __bf16* bl = (const __bf16*)Wlo + ((size_t)(r * 8 + w) * 4) * 512 + lane * 16;

    v8f acc = {};
#pragma unroll
    for (int c = 0; c < 4; ++c) {
        v16bf ahi = *(const v16bf*)(ah + (size_t)c * 512);
        v16bf alo = *(const v16bf*)(al + (size_t)c * 512);
        v16bf bhi = *(const v16bf*)(bh + (size_t)c * 512);
        v16bf blo = *(const v16bf*)(bl + (size_t)c * 512);
        acc = __builtin_amdgcn_wmma_f32_16x16x32_bf16(false, ahi, false, bhi,
                                                      (short)0, acc, false, false);
        acc = __builtin_amdgcn_wmma_f32_16x16x32_bf16(false, ahi, false, blo,
                                                      (short)0, acc, false, false);
        acc = __builtin_amdgcn_wmma_f32_16x16x32_bf16(false, alo, false, bhi,
                                                      (short)0, acc, false, false);
    }

    // C/D layout: lane L: n = L%16 ; VGPR v -> m = v + 8*(L/16)
    const int n     = lane & 15;
    const int mbase = (lane >> 4) * 8;
#pragma unroll
    for (int v = 0; v < 8; ++v) {
        int m = mt * 16 + mbase + v;
        if (m < n_nodes)
            HR[((size_t)r * n_nodes + m) * FEATS + w * 16 + n] = acc[v];
    }
}

// ---------------------------------------------------------------------------
// 2b) FALLBACK GEMM (small workspace): A split computed inline per wave.
// ---------------------------------------------------------------------------
__global__ void __launch_bounds__(256)
rgcn_gemm(const float* __restrict__ X,
          const unsigned short* __restrict__ Whi,
          const unsigned short* __restrict__ Wlo,
          float* __restrict__ HR, int n_nodes) {
    const int lane = threadIdx.x & 31;
    const int w    = threadIdx.x >> 5;
    const int r    = blockIdx.y;
    const int m0   = blockIdx.x * 16;

    const int mloc = lane & 15;
    const int kg   = lane >> 4;
    int row = m0 + mloc;
    if (row >= n_nodes) row = n_nodes - 1;      // clamp: keeps EXEC all-ones
    const float* xr = X + (size_t)row * FEATS;

    const __bf16* bh = (const __bf16*)Whi + ((size_t)(r * 8 + w) * 4) * 512 + lane * 16;
    const __bf16* bl = (const __bf16*)Wlo + ((size_t)(r * 8 + w) * 4) * 512 + lane * 16;

    v8f acc = {};
#pragma unroll
    for (int c = 0; c < 4; ++c) {
        v16bf ahi, alo;
#pragma unroll
        for (int h = 0; h < 16; ++h) {
            int K = c * 32 + (h >> 3) * 16 + kg * 8 + (h & 7);
            float x = xr[K];
            __bf16 hi = (__bf16)x;
            ahi[h] = hi;
            alo[h] = (__bf16)(x - (float)hi);
        }
        v16bf bhi = *(const v16bf*)(bh + (size_t)c * 512);
        v16bf blo = *(const v16bf*)(bl + (size_t)c * 512);
        acc = __builtin_amdgcn_wmma_f32_16x16x32_bf16(false, ahi, false, bhi,
                                                      (short)0, acc, false, false);
        acc = __builtin_amdgcn_wmma_f32_16x16x32_bf16(false, ahi, false, blo,
                                                      (short)0, acc, false, false);
        acc = __builtin_amdgcn_wmma_f32_16x16x32_bf16(false, alo, false, bhi,
                                                      (short)0, acc, false, false);
    }

    const int n     = lane & 15;
    const int mbase = (lane >> 4) * 8;
#pragma unroll
    for (int v = 0; v < 8; ++v) {
        int m = m0 + mbase + v;
        if (m < n_nodes)
            HR[((size_t)r * n_nodes + m) * FEATS + w * 16 + n] = acc[v];
    }
}

// ---------------------------------------------------------------------------
// 3) per-edge gather * norm, scatter-add into dst. One wave per edge,
//    lane handles 4 consecutive feats (32*4 = 128). HR (76.8 MB) lives in L2.
// ---------------------------------------------------------------------------
__global__ void __launch_bounds__(256)
rgcn_edges(const float* __restrict__ HR, const float* __restrict__ norm,
           const int* __restrict__ src, const int* __restrict__ dst,
           const int* __restrict__ rel, float* __restrict__ out,
           int n_edges, int n_nodes) {
    int e    = blockIdx.x * 8 + (threadIdx.x >> 5);
    int lane = threadIdx.x & 31;
    if (e >= n_edges) return;                   // wave-uniform, no divergence
    int   s  = src[e];
    int   d  = dst[e];
    int   r  = rel[e];
    float nv = norm[e];
    const float4* rowp = (const float4*)(HR + ((size_t)r * n_nodes + s) * FEATS);
    float4 v = rowp[lane];
    float* op = out + (size_t)d * FEATS + lane * 4;
    __hip_atomic_fetch_add(op + 0, v.x * nv, __ATOMIC_RELAXED, __HIP_MEMORY_SCOPE_AGENT);
    __hip_atomic_fetch_add(op + 1, v.y * nv, __ATOMIC_RELAXED, __HIP_MEMORY_SCOPE_AGENT);
    __hip_atomic_fetch_add(op + 2, v.z * nv, __ATOMIC_RELAXED, __HIP_MEMORY_SCOPE_AGENT);
    __hip_atomic_fetch_add(op + 3, v.w * nv, __ATOMIC_RELAXED, __HIP_MEMORY_SCOPE_AGENT);
}

// ---------------------------------------------------------------------------
// 4) ReLU in place
// ---------------------------------------------------------------------------
__global__ void rgcn_relu(float* __restrict__ p, int n) {
    int i = blockIdx.x * blockDim.x + threadIdx.x;
    if (i < n) p[i] = fmaxf(p[i], 0.f);
}

// ---------------------------------------------------------------------------
extern "C" void kernel_launch(void* const* d_in, const int* in_sizes, int n_in,
                              void* d_out, int out_size, void* d_ws, size_t ws_size,
                              hipStream_t stream) {
    const float* feature = (const float*)d_in[0];   // [N, 128] f32
    const float* weight  = (const float*)d_in[1];   // [3, 128, 128] f32
    const float* norm    = (const float*)d_in[2];   // [E] f32
    const int*   src     = (const int*)d_in[3];     // [E] i32
    const int*   dst     = (const int*)d_in[4];     // [E] i32
    const int*   rel     = (const int*)d_in[5];     // [E] i32
    float*       out     = (float*)d_out;           // [N, 128] f32

    const int n_nodes = in_sizes[0] / FEATS;
    const int n_edges = in_sizes[2];
    const int mtiles  = (n_nodes + 15) / 16;

    // workspace: HR [3,N,128] f32 | Whi,Wlo bf16 frags | Xhi,Xlo bf16 frags
    float* HR = (float*)d_ws;
    size_t hr_bytes = (size_t)N_REL * n_nodes * FEATS * sizeof(float);  // 76.8 MB
    hr_bytes = (hr_bytes + 255) & ~(size_t)255;
    size_t wfrag_elems = (size_t)N_REL * 8 * 4 * 32 * 16;               // 49152
    size_t xfrag_elems = (size_t)mtiles * 4 * 32 * 16;                  // N*128
    unsigned short* Whi = (unsigned short*)((char*)d_ws + hr_bytes);
    unsigned short* Wlo = Whi + wfrag_elems;
    unsigned short* Xhi = Wlo + wfrag_elems;
    unsigned short* Xlo = Xhi + xfrag_elems;
    size_t need_fast = hr_bytes + 2 * wfrag_elems * 2 + 2 * xfrag_elems * 2;

    int n_out  = n_nodes * FEATS;
    int n_out4 = n_out / 4;
    rgcn_zero_out<<<(n_out4 + 255) / 256, 256, 0, stream>>>((float4*)out, n_out4);

    int n_w = N_REL * FEATS * FEATS;
    rgcn_prep_w<<<(n_w + 255) / 256, 256, 0, stream>>>(weight, Whi, Wlo);

    dim3 grid(mtiles, N_REL);
    if (ws_size >= need_fast) {
        int n_x = n_nodes * FEATS;
        rgcn_prep_x<<<(n_x + 255) / 256, 256, 0, stream>>>(feature, Xhi, Xlo, n_x);
        rgcn_gemm_pre<<<grid, 256, 0, stream>>>(Xhi, Xlo, Whi, Wlo, HR, n_nodes);
    } else {
        rgcn_gemm<<<grid, 256, 0, stream>>>(feature, Whi, Wlo, HR, n_nodes);
    }

    rgcn_edges<<<(n_edges + 7) / 8, 256, 0, stream>>>(HR, norm, src, dst, rel,
                                                      out, n_edges, n_nodes);

    rgcn_relu<<<(n_out + 255) / 256, 256, 0, stream>>>(out, n_out);
}
